// LSTM_24369644437986
// MI455X (gfx1250) — matrix operand order repaired and checked by reading, not compiled, not run on hardware
//
#include <hip/hip_runtime.h>
#include <hip/hip_bf16.h>

// LSTM: B=64, S=512, D=512, H=1024
#define LSTM_B 64
#define LSTM_S 512
#define LSTM_D 512
#define LSTM_H 1024
#define LSTM_KT (LSTM_D + LSTM_H)   // 1536 : fused K (input + recurrent)
#define LSTM_G4H (4 * LSTM_H)       // 4096 : gate dimension

typedef __attribute__((ext_vector_type(16))) _Float16 v16h;
typedef __attribute__((ext_vector_type(8)))  _Float16 v8h;
typedef __attribute__((ext_vector_type(8)))  float    v8f;

// ---------------------------------------------------------------------------
// Convert inputs to f16, build fused weight matrix Wc[g][0:512]=Wx, [512:1536]=Wh,
// and fold biases bsum = bx + bh.
// ---------------------------------------------------------------------------
__global__ __launch_bounds__(256) void lstm_convert(
    const float* __restrict__ x, const float* __restrict__ Wx,
    const float* __restrict__ Wh, const float* __restrict__ bx,
    const float* __restrict__ bh, _Float16* __restrict__ xh,
    _Float16* __restrict__ Wc, float* __restrict__ bsum) {
  const size_t tid    = (size_t)blockIdx.x * blockDim.x + threadIdx.x;
  const size_t stride = (size_t)gridDim.x * blockDim.x;

  const size_t nx = (size_t)LSTM_B * LSTM_S * LSTM_D;
  for (size_t i = tid; i < nx; i += stride) xh[i] = (_Float16)x[i];

  const size_t nw = (size_t)LSTM_G4H * LSTM_KT;
  for (size_t j = tid; j < nw; j += stride) {
    const size_t g = j / LSTM_KT;
    const size_t k = j - g * LSTM_KT;
    const float  w = (k < LSTM_D) ? Wx[g * LSTM_D + k]
                                  : Wh[g * LSTM_H + (k - LSTM_D)];
    Wc[j] = (_Float16)w;
  }

  for (size_t n = tid; n < (size_t)LSTM_G4H; n += stride) bsum[n] = bx[n] + bh[n];
}

// ---------------------------------------------------------------------------
// Zero initial hidden/cell state (both h ping-pong buffers).
// ---------------------------------------------------------------------------
__global__ __launch_bounds__(256) void lstm_init(_Float16* __restrict__ h0,
                                                 _Float16* __restrict__ h1,
                                                 float* __restrict__ c) {
  const int i = blockIdx.x * blockDim.x + threadIdx.x;
  if (i < LSTM_B * LSTM_H) {
    h0[i] = (_Float16)0.0f;
    h1[i] = (_Float16)0.0f;
    c[i]  = 0.0f;
  }
}

// Hard scheduling fence: nothing may be reordered across it.
#define LSTM_SB() __builtin_amdgcn_sched_barrier(0)

// Load one 32-wide K chunk: A fragment + 4 gate B fragments.
#define LSTM_LOAD_CHUNK(KB, A, B0, B1, B2, B3)                                 \
  {                                                                            \
    const v8h alo_ = *(const v8h*)(abase + (KB) + 16 * lhi);                   \
    const v8h ahi_ = *(const v8h*)(abase + (KB) + 8 + 16 * lhi);               \
    (A) = __builtin_shufflevector(alo_, ahi_, 0, 1, 2, 3, 4, 5, 6, 7, 8, 9,    \
                                  10, 11, 12, 13, 14, 15);                     \
    (B0) = *(const v16h*)(w0 + (KB));                                          \
    (B1) = *(const v16h*)(w1 + (KB));                                          \
    (B2) = *(const v16h*)(w2 + (KB));                                          \
    (B3) = *(const v16h*)(w3 + (KB));                                          \
  }

// Issue the 4 gate WMMAs for one chunk.
#define LSTM_WMMA4(A, B0, B1, B2, B3)                                          \
  acc0 = __builtin_amdgcn_wmma_f32_16x16x32_f16(false, (A), false, (B0),       \
                                                (short)0, acc0, false, false); \
  acc1 = __builtin_amdgcn_wmma_f32_16x16x32_f16(false, (A), false, (B1),       \
                                                (short)0, acc1, false, false); \
  acc2 = __builtin_amdgcn_wmma_f32_16x16x32_f16(false, (A), false, (B2),       \
                                                (short)0, acc2, false, false); \
  acc3 = __builtin_amdgcn_wmma_f32_16x16x32_f16(false, (A), false, (B3),       \
                                                (short)0, acc3, false, false)

// ---------------------------------------------------------------------------
// One fused LSTM time step, split-K across 6 waves per tile.
//   Block = 192 threads (6 waves), one 16x16 (m,n) block of H per block,
//   computing all 4 gate tiles (i/f/g/o) that share the same A fragments.
//   Wave 0..1 : x-region, wave 2..5 : h-region; 8 chunks (K=256) per wave.
//   The 8 chunks run as a straight-line two-deep software pipeline with two
//   register fragment sets, pinned by sched_barriers so the next chunk's 10
//   loads stay in flight across each WMMA group (partial s_wait_loadcnt).
//   Partial sums -> LDS -> barrier -> cross-wave reduce + gate pointwise.
// Grid: 256 blocks = (4 m-tiles) x (64 n-tiles).
// ---------------------------------------------------------------------------
__global__ __launch_bounds__(192) void lstm_step(
    const _Float16* __restrict__ xh,   // (B,S,D) f16
    const _Float16* __restrict__ Wc,   // (4H, KT) f16
    const float* __restrict__ bsum,    // (4H) f32
    const _Float16* __restrict__ hr,   // h_{s-1} (B,H) f16  (read)
    _Float16* __restrict__ hw,         // h_s     (B,H) f16  (write)
    float* __restrict__ cst,           // c state (B,H) f32  (in/out)
    float* __restrict__ out,           // hidden (B,S,H) ++ hT (B,H) ++ cT (B,H)
    int s) {
  __shared__ __align__(32) float red[6 * 4 * 32 * 8];  // 24 KB partials

  const int lane = threadIdx.x & 31;
  const int lrow = lane & 15;            // M (A) or N (B/C/D) within tile
  const int lhi  = lane >> 4;            // lane half
  const int wave = threadIdx.x >> 5;     // 0..5 split-K slice
  const int m0   = (blockIdx.x & 3) << 4;
  const int n0   = (blockIdx.x >> 2) << 4;
  const int n    = n0 + lrow;

  // Per-wave K slice: single region, fixed base pointers.
  const _Float16* abase;
  int kstart;
  if (wave < 2) {
    kstart = wave * 256;
    abase  = xh + (size_t)(m0 + lrow) * (LSTM_S * LSTM_D) +
             (size_t)s * LSTM_D + kstart;
  } else {
    kstart = LSTM_D + (wave - 2) * 256;
    abase  = hr + (size_t)(m0 + lrow) * LSTM_H + (kstart - LSTM_D);
  }
  const _Float16* w0 = Wc + (size_t)(0 * LSTM_H + n) * LSTM_KT + kstart + 16 * lhi;
  const _Float16* w1 = Wc + (size_t)(1 * LSTM_H + n) * LSTM_KT + kstart + 16 * lhi;
  const _Float16* w2 = Wc + (size_t)(2 * LSTM_H + n) * LSTM_KT + kstart + 16 * lhi;
  const _Float16* w3 = Wc + (size_t)(3 * LSTM_H + n) * LSTM_KT + kstart + 16 * lhi;

  v8f acc0 = {0.f, 0.f, 0.f, 0.f, 0.f, 0.f, 0.f, 0.f};
  v8f acc1 = acc0, acc2 = acc0, acc3 = acc0;

  // Two-deep software pipeline over 8 chunks (K offsets 0,32,...,224),
  // pinned with scheduling fences at every stage boundary.
  v16h a0, b00, b01, b02, b03;
  v16h a1, b10, b11, b12, b13;

  LSTM_LOAD_CHUNK(0,   a0, b00, b01, b02, b03);   // preload chunk 0 -> S0
  LSTM_LOAD_CHUNK(32,  a1, b10, b11, b12, b13);   // preload chunk 1 -> S1
  LSTM_SB();
  LSTM_WMMA4(a0, b00, b01, b02, b03);             // chunk 0 (chunk 1 in flight)
  LSTM_SB();
  LSTM_LOAD_CHUNK(64,  a0, b00, b01, b02, b03);   // chunk 2 -> S0
  LSTM_SB();
  LSTM_WMMA4(a1, b10, b11, b12, b13);             // chunk 1 (chunk 2 in flight)
  LSTM_SB();
  LSTM_LOAD_CHUNK(96,  a1, b10, b11, b12, b13);   // chunk 3 -> S1
  LSTM_SB();
  LSTM_WMMA4(a0, b00, b01, b02, b03);             // chunk 2 (chunk 3 in flight)
  LSTM_SB();
  LSTM_LOAD_CHUNK(128, a0, b00, b01, b02, b03);   // chunk 4 -> S0
  LSTM_SB();
  LSTM_WMMA4(a1, b10, b11, b12, b13);             // chunk 3 (chunk 4 in flight)
  LSTM_SB();
  LSTM_LOAD_CHUNK(160, a1, b10, b11, b12, b13);   // chunk 5 -> S1
  LSTM_SB();
  LSTM_WMMA4(a0, b00, b01, b02, b03);             // chunk 4 (chunk 5 in flight)
  LSTM_SB();
  LSTM_LOAD_CHUNK(192, a0, b00, b01, b02, b03);   // chunk 6 -> S0
  LSTM_SB();
  LSTM_WMMA4(a1, b10, b11, b12, b13);             // chunk 5 (chunk 6 in flight)
  LSTM_SB();
  LSTM_LOAD_CHUNK(224, a1, b10, b11, b12, b13);   // chunk 7 -> S1
  LSTM_SB();
  LSTM_WMMA4(a0, b00, b01, b02, b03);             // chunk 6 (chunk 7 in flight)
  LSTM_SB();
  LSTM_WMMA4(a1, b10, b11, b12, b13);             // chunk 7

  // Spill partials to LDS in register layout: ((wave*4+g)*32 + lane)*8 + v
  {
    const int base = ((wave * 4 + 0) * 32 + lane) * 8;
    *(v8f*)&red[base]          = acc0;
    *(v8f*)&red[base + 32 * 8] = acc1;  // gate stride = 32*8 floats
    *(v8f*)&red[base + 64 * 8] = acc2;
    *(v8f*)&red[base + 96 * 8] = acc3;
  }
  __syncthreads();

  // Cross-wave reduce + gate pointwise. 256 tile elements over 192 threads.
  // Element e: lane_e = e>>3, v_e = e&7 -> m_rel = v_e + 8*(lane_e>>4),
  //                                         n_rel = lane_e & 15.
  for (int e = threadIdx.x; e < 256; e += 192) {
    const int lane_e = e >> 3;
    const int v_e    = e & 7;
    const int m      = m0 + v_e + ((lane_e >> 4) << 3);
    const int nn     = n0 + (lane_e & 15);
    const int off    = lane_e * 8 + v_e;

    float g0 = bsum[0 * LSTM_H + nn];
    float g1 = bsum[1 * LSTM_H + nn];
    float g2 = bsum[2 * LSTM_H + nn];
    float g3 = bsum[3 * LSTM_H + nn];
#pragma unroll
    for (int w = 0; w < 6; ++w) {
      const int wb = w * 4 * 32 * 8;
      g0 += red[wb + 0 * 32 * 8 + off];
      g1 += red[wb + 1 * 32 * 8 + off];
      g2 += red[wb + 2 * 32 * 8 + off];
      g3 += red[wb + 3 * 32 * 8 + off];
    }

    const float iv = 1.0f / (1.0f + __expf(-g0));
    const float fv = 1.0f / (1.0f + __expf(-g1));
    const float gv = tanhf(g2);
    const float ov = 1.0f / (1.0f + __expf(-g3));

    const size_t ci = (size_t)m * LSTM_H + nn;
    const float cn = fv * cst[ci] + iv * gv;
    cst[ci] = cn;
    const float hn = ov * tanhf(cn);
    hw[ci] = (_Float16)hn;  // feeds next step's A operand
    out[(size_t)m * (LSTM_S * LSTM_H) + (size_t)s * LSTM_H + nn] = hn;
    if (s == LSTM_S - 1) {  // final (h_T, c_T) outputs
      const size_t tail = (size_t)LSTM_B * LSTM_S * LSTM_H;
      out[tail + ci] = hn;
      out[tail + (size_t)LSTM_B * LSTM_H + ci] = cn;
    }
  }
}

// ---------------------------------------------------------------------------
// Host launcher. Workspace layout (bytes):
//   xh   : 0         (B*S*D f16 = 33,554,432)
//   Wc   : 33554432  (4H*KT f16 = 12,582,912)
//   bsum : 46137344  (4H f32 = 16,384)
//   hh0  : 46153728  (B*H f16 = 131,072)
//   hh1  : 46284800  (B*H f16 = 131,072)
//   c    : 46415872  (B*H f32 = 262,144)   total ~44.5 MB
// ---------------------------------------------------------------------------
extern "C" void kernel_launch(void* const* d_in, const int* in_sizes, int n_in,
                              void* d_out, int out_size, void* d_ws,
                              size_t ws_size, hipStream_t stream) {
  const float* x  = (const float*)d_in[0];
  const float* Wx = (const float*)d_in[1];
  const float* Wh = (const float*)d_in[2];
  const float* bx = (const float*)d_in[3];
  const float* bh = (const float*)d_in[4];
  float* out = (float*)d_out;

  char* ws = (char*)d_ws;
  _Float16* xh   = (_Float16*)(ws);
  _Float16* Wc   = (_Float16*)(ws + 33554432);
  float*    bsum = (float*)(ws + 46137344);
  _Float16* hh0  = (_Float16*)(ws + 46153728);
  _Float16* hh1  = (_Float16*)(ws + 46284800);
  float*    cst  = (float*)(ws + 46415872);

  lstm_convert<<<2048, 256, 0, stream>>>(x, Wx, Wh, bx, bh, xh, Wc, bsum);
  lstm_init<<<(LSTM_B * LSTM_H + 255) / 256, 256, 0, stream>>>(hh0, hh1, cst);

  for (int s = 0; s < LSTM_S; ++s) {
    const _Float16* hr = (s & 1) ? hh1 : hh0;
    _Float16*       hw = (s & 1) ? hh0 : hh1;
    lstm_step<<<256, 192, 0, stream>>>(xh, Wc, bsum, hr, hw, cst, out, s);
  }
}
